// LinearAttention_80041010528410
// MI455X (gfx1250) — matrix operand order repaired and checked by reading, not compiled, not run on hardware
//
#include <hip/hip_runtime.h>

typedef __attribute__((ext_vector_type(16))) _Float16 v16h;
typedef __attribute__((ext_vector_type(8)))  _Float16 v8h;
typedef __attribute__((ext_vector_type(4)))  _Float16 v4h;
typedef __attribute__((ext_vector_type(8)))  float    v8f;
typedef __attribute__((ext_vector_type(4)))  float    v4f;

#define HEADS   8
#define DH      32
#define DMODEL  256
#define NTOK    131072              // B*N = 8*16384
#define QK_SCALE 0.17677669529663687f   // 32^-0.5
#define LN_EPS  1e-5f

__device__ __forceinline__ v8f wmma16(v16h a, v16h b, v8f c) {
  return __builtin_amdgcn_wmma_f32_16x16x32_f16(false, a, false, b, (short)0, c, false, false);
}
__device__ __forceinline__ v16h cat8(v8h lo, v8h hi) {
  return __builtin_shufflevector(lo, hi, 0,1,2,3,4,5,6,7,8,9,10,11,12,13,14,15);
}
__device__ __forceinline__ v8f z8() {
  v8f z;
#pragma unroll
  for (int i = 0; i < 8; ++i) z[i] = 0.f;
  return z;
}

// Compute one 16x32 x 256 -> two 16x16 tiles (cols colbase, colbase+16).
// A from LDS (padded stride 264), B from global f16 W^T (stride 256).
__device__ __forceinline__ void gemm_tile2(const _Float16* __restrict__ xs,
                                           const _Float16* __restrict__ WT,
                                           int colbase, int mb, int l16, int hi,
                                           v8f& acc0, v8f& acc1) {
  acc0 = z8(); acc1 = z8();
  const int arow = mb + l16;
#pragma unroll 2
  for (int k0 = 0; k0 < 256; k0 += 32) {
    const _Float16* ap = &xs[arow * 264 + k0 + hi * 16];
    v16h A = cat8(*(const v8h*)ap, *(const v8h*)(ap + 8));
    const _Float16* bp = &WT[(colbase + l16) * 256 + k0 + hi * 16];
    v16h B0 = cat8(*(const v8h*)bp,             *(const v8h*)(bp + 8));
    v16h B1 = cat8(*(const v8h*)(bp + 16*256),  *(const v8h*)(bp + 16*256 + 8));
    acc0 = wmma16(A, B0, acc0);
    acc1 = wmma16(A, B1, acc1);
  }
}

// ---------------- Kernel 1: prep (Wqkv -> f16 transposed, zero accumulators) ----
__global__ void k_prep(const float* __restrict__ Wqkv, _Float16* __restrict__ WT,
                       float* __restrict__ ctx, float* __restrict__ cs) {
  int t = blockIdx.x * 256 + threadIdx.x;
  if (t < 768 * 256) {
    int n = t >> 8, k = t & 255;
    WT[t] = (_Float16)Wqkv[k * 768 + n];     // WT[n][k] = Wqkv[k][n]
  }
  if (t < 64 * 32 * 32) ctx[t] = 0.f;
  if (t < 64 * 32)      cs[t]  = 0.f;
}

// ---------------- Kernel 2: x@Wqkv + q-softmax + exp(k) + context accumulation --
__global__ void __launch_bounds__(256)
k_gemm1(const float* __restrict__ x, const _Float16* __restrict__ WT,
        _Float16* __restrict__ qws, float* __restrict__ ctx, float* __restrict__ cs) {
  __shared__ _Float16 xs[32 * 264];               // x tile, f16, padded stride
  __shared__ _Float16 tkv[HEADS][2][32 * 40];     // per-head transposed ek / v

  const int tid  = threadIdx.x;
  const int lane = tid & 31;
  const int l16  = lane & 15;
  const int hi   = lane >> 4;
  const int h    = tid >> 5;                      // wave id == head
  const int bh   = (blockIdx.x >> 6) * HEADS + h; // 64 blocks per batch

  _Float16* kt = &tkv[h][0][0];
  _Float16* vt = &tkv[h][1][0];

  v8f cacc[4];
#pragma unroll
  for (int i = 0; i < 4; ++i) cacc[i] = z8();
  float csum0 = 0.f, csum1 = 0.f;

#pragma unroll 1
  for (int iter = 0; iter < 8; ++iter) {
    const int row0 = blockIdx.x * 256 + iter * 32;
    __syncthreads();
    // ---- stage x tile (32 x 256 fp32) -> f16 LDS, coalesced, non-temporal ----
    {
      const v4f* xg = reinterpret_cast<const v4f*>(x + (size_t)row0 * DMODEL);
#pragma unroll
      for (int j = 0; j < 8; ++j) {
        int f = tid + j * 256;                    // float4 index 0..2047
        v4f vv = __builtin_nontemporal_load(&xg[f]);
        int r = f >> 6;
        int c = (f & 63) << 2;
        v4h hv;
        hv[0] = (_Float16)vv[0]; hv[1] = (_Float16)vv[1];
        hv[2] = (_Float16)vv[2]; hv[3] = (_Float16)vv[3];
        *reinterpret_cast<v4h*>(&xs[r * 264 + c]) = hv;
      }
    }
    __syncthreads();

    // ---- per-wave: section passes q, k, v (2 tiles each) for 2 row tiles ----
#pragma unroll 1
    for (int mb = 0; mb <= 16; mb += 16) {
      v8f a0, a1;

      // ---- q section: softmax over 32 features, *SCALE, store f16 ----
      gemm_tile2(xs, WT, 0 * 256 + h * 32, mb, l16, hi, a0, a1);
#pragma unroll
      for (int r = 0; r < 8; ++r) {
        float v0 = a0[r], v1 = a1[r];
        float m = fmaxf(v0, v1);
        m = fmaxf(m, __shfl_xor(m, 1)); m = fmaxf(m, __shfl_xor(m, 2));
        m = fmaxf(m, __shfl_xor(m, 4)); m = fmaxf(m, __shfl_xor(m, 8));
        float e0 = __expf(v0 - m), e1 = __expf(v1 - m);
        float s = e0 + e1;
        s += __shfl_xor(s, 1); s += __shfl_xor(s, 2);
        s += __shfl_xor(s, 4); s += __shfl_xor(s, 8);
        float inv = QK_SCALE / s;
        int grow = row0 + mb + (hi << 3) + r;
        qws[(size_t)grow * 256 + h * 32 + l16]      = (_Float16)(e0 * inv);
        qws[(size_t)grow * 256 + h * 32 + 16 + l16] = (_Float16)(e1 * inv);
      }

      // ---- k section: exp, column-sum accumulation, transposed store ----
      gemm_tile2(xs, WT, 1 * 256 + h * 32, mb, l16, hi, a0, a1);
      {
        v8h ek0h, ek1h;
        float s0 = 0.f, s1 = 0.f;
#pragma unroll
        for (int r = 0; r < 8; ++r) {
          float e0 = __expf(a0[r]); float e1 = __expf(a1[r]);
          s0 += e0; s1 += e1;
          ek0h[r] = (_Float16)e0; ek1h[r] = (_Float16)e1;
        }
        csum0 += s0; csum1 += s1;
        *reinterpret_cast<v8h*>(&kt[(l16)      * 40 + mb + hi * 8]) = ek0h;
        *reinterpret_cast<v8h*>(&kt[(16 + l16) * 40 + mb + hi * 8]) = ek1h;
      }

      // ---- v section: transposed store ----
      gemm_tile2(xs, WT, 2 * 256 + h * 32, mb, l16, hi, a0, a1);
      {
        v8h v0h, v1h;
#pragma unroll
        for (int r = 0; r < 8; ++r) { v0h[r] = (_Float16)a0[r]; v1h[r] = (_Float16)a1[r]; }
        *reinterpret_cast<v8h*>(&vt[(l16)      * 40 + mb + hi * 8]) = v0h;
        *reinterpret_cast<v8h*>(&vt[(16 + l16) * 40 + mb + hi * 8]) = v1h;
      }
    }

    // ---- context: ctx += ek^T (32d x 32n) @ v (32n x 32e), in-register accum ----
    asm volatile("s_wait_dscnt 0" ::: "memory");
#pragma unroll
    for (int db = 0; db < 2; ++db) {
      const _Float16* ap = &kt[(db * 16 + l16) * 40 + hi * 16];
      v16h A = cat8(*(const v8h*)ap, *(const v8h*)(ap + 8));
#pragma unroll
      for (int eb = 0; eb < 2; ++eb) {
        const _Float16* bp = &vt[(eb * 16 + l16) * 40 + hi * 16];
        v16h B = cat8(*(const v8h*)bp, *(const v8h*)(bp + 8));
        cacc[db * 2 + eb] = wmma16(A, B, cacc[db * 2 + eb]);
      }
    }
  }

  // ---- flush per-block partials ----
  csum0 += __shfl_xor(csum0, 16);
  csum1 += __shfl_xor(csum1, 16);
  if (lane < 16) {
    atomicAdd(&cs[bh * 32 + l16],      csum0);
    atomicAdd(&cs[bh * 32 + 16 + l16], csum1);
  }
#pragma unroll
  for (int db = 0; db < 2; ++db)
#pragma unroll
    for (int eb = 0; eb < 2; ++eb)
#pragma unroll
      for (int r = 0; r < 8; ++r) {
        int d = db * 16 + hi * 8 + r;
        int e = eb * 16 + l16;
        atomicAdd(&ctx[((size_t)bh * 32 + d) * 32 + e], cacc[db * 2 + eb][r]);
      }
}

// ---------------- Kernel 3: fold (ctx/colsum) @ Wout into per-batch M^T (f16) ----
__global__ void k_buildM(const float* __restrict__ ctx, const float* __restrict__ cs,
                         const float* __restrict__ Wout, _Float16* __restrict__ Mt) {
  const int b  = blockIdx.x >> 5;
  const int eg = (blockIdx.x & 31) * 8;
  const int hd = threadIdx.x;
  const int h = hd >> 5, d = hd & 31;
  const float inv = 1.f / cs[(b * 8 + h) * 32 + d];
  float c[32];
#pragma unroll
  for (int e = 0; e < 32; ++e)
    c[e] = ctx[(((size_t)b * 8 + h) * 32 + d) * 32 + e] * inv;
  for (int ep = eg; ep < eg + 8; ++ep) {
    float acc = 0.f;
#pragma unroll
    for (int e = 0; e < 32; ++e) acc += c[e] * Wout[(h * 32 + e) * 256 + ep];
    Mt[((size_t)b * 256 + ep) * 256 + hd] = (_Float16)acc;   // Mt[b][e'][hd]
  }
}

// ---------------- Kernel 4: out = q @ M, fused gamma-LayerNorm ------------------
__global__ void __launch_bounds__(256)
k_gemm2(const _Float16* __restrict__ q, const _Float16* __restrict__ Mt,
        const float* __restrict__ gamma, float* __restrict__ out) {
  __shared__ _Float16 qs[32 * 264];               // staged q tile
  __shared__ float os[32 * 268];                  // fp32 result staging
  const int tid = threadIdx.x, lane = tid & 31, l16 = lane & 15, hi = lane >> 4, w = tid >> 5;
  const int row0 = blockIdx.x * 32;
  const _Float16* MB = Mt + (size_t)(blockIdx.x >> 9) * 65536;  // 512 blocks / batch

  // ---- stage q tile (32 x 256 f16) once per block, coalesced ----
  {
    const v8h* qg = reinterpret_cast<const v8h*>(q + (size_t)row0 * 256);
#pragma unroll
    for (int j = 0; j < 4; ++j) {
      int f = tid + j * 256;                      // v8h index 0..1023
      v8h vv = qg[f];
      int r = f >> 5;
      int c = (f & 31) << 3;
      *reinterpret_cast<v8h*>(&qs[r * 264 + c]) = vv;
    }
  }
  __syncthreads();

#pragma unroll 1
  for (int mb = 0; mb <= 16; mb += 16) {
    v8f a0 = z8(), a1 = z8();
#pragma unroll 2
    for (int k0 = 0; k0 < 256; k0 += 32) {
      const _Float16* ap = &qs[(mb + l16) * 264 + k0 + hi * 16];
      v16h A = cat8(*(const v8h*)ap, *(const v8h*)(ap + 8));
      const _Float16* b0p = &MB[(w * 32 + l16) * 256 + k0 + hi * 16];
      const _Float16* b1p = b0p + 16 * 256;
      v16h B0 = cat8(*(const v8h*)b0p, *(const v8h*)(b0p + 8));
      v16h B1 = cat8(*(const v8h*)b1p, *(const v8h*)(b1p + 8));
      a0 = wmma16(A, B0, a0);
      a1 = wmma16(A, B1, a1);
    }
#pragma unroll
    for (int r = 0; r < 8; ++r) {
      int rr = mb + (hi << 3) + r;
      os[rr * 268 + w * 32 + l16]      = a0[r];
      os[rr * 268 + w * 32 + 16 + l16] = a1[r];
    }
  }
  __syncthreads();

  // LayerNorm: 8 threads per row, 32 cols each
  const int row = tid >> 3;
  const int c0  = (tid & 7) * 32;
  float s = 0.f, ss = 0.f;
#pragma unroll
  for (int j = 0; j < 32; j += 4) {
    v4f vv = *(const v4f*)&os[row * 268 + c0 + j];
    s  += vv[0] + vv[1] + vv[2] + vv[3];
    ss += vv[0]*vv[0] + vv[1]*vv[1] + vv[2]*vv[2] + vv[3]*vv[3];
  }
  s  += __shfl_xor(s, 1);  s  += __shfl_xor(s, 2);  s  += __shfl_xor(s, 4);
  ss += __shfl_xor(ss, 1); ss += __shfl_xor(ss, 2); ss += __shfl_xor(ss, 4);
  const float mu  = s * (1.f / 256.f);
  const float var = ss * (1.f / 256.f) - mu * mu;
  const float ri  = rsqrtf(var + LN_EPS);
  float* op = out + (size_t)(row0 + row) * 256 + c0;
#pragma unroll
  for (int j = 0; j < 32; j += 4) {
    v4f g  = *(const v4f*)&gamma[c0 + j];
    v4f vv = *(const v4f*)&os[row * 268 + c0 + j];
    v4f o;
    o[0] = (vv[0] - mu) * ri * g[0]; o[1] = (vv[1] - mu) * ri * g[1];
    o[2] = (vv[2] - mu) * ri * g[2]; o[3] = (vv[3] - mu) * ri * g[3];
    __builtin_nontemporal_store(o, (v4f*)&op[j]);
  }
}

// ---------------- launcher ------------------------------------------------------
extern "C" void kernel_launch(void* const* d_in, const int* in_sizes, int n_in,
                              void* d_out, int out_size, void* d_ws, size_t ws_size,
                              hipStream_t stream) {
  (void)in_sizes; (void)n_in; (void)out_size; (void)ws_size;
  const float* x    = (const float*)d_in[0];
  const float* Wqkv = (const float*)d_in[1];
  const float* Wout = (const float*)d_in[2];
  const float* gam  = (const float*)d_in[3];
  float* outp = (float*)d_out;

  char* ws = (char*)d_ws;
  _Float16* WT  = (_Float16*)(ws);                       // 768*256*2   = 393216
  _Float16* qws = (_Float16*)(ws + 393216);              // 131072*256*2= 67108864
  float*    ctx = (float*)(ws + 67502080);               // 64*32*32*4  = 262144
  float*    cs  = (float*)(ws + 67764224);               // 64*32*4     = 8192
  _Float16* Mt  = (_Float16*)(ws + 67772416);            // 8*256*256*2 = 1048576

  k_prep  <<<768,  256, 0, stream>>>(Wqkv, WT, ctx, cs);
  k_gemm1 <<<512,  256, 0, stream>>>(x, WT, qws, ctx, cs);
  k_buildM<<<256,  256, 0, stream>>>(ctx, cs, Wout, Mt);
  k_gemm2 <<<4096, 256, 0, stream>>>(qws, Mt, gam, outp);
}
